// Net_ra_58248346468974
// MI455X (gfx1250) — compile-verified, matrix operand order checked by reading, not run on hardware
//
#include <hip/hip_runtime.h>
#include <stdint.h>

#define L_SEQ 2048
#define BATCH 4
#define CDIM  1024
#define NHEAD 16
#define DHEAD 64
#define WSZ   11
#define PADW  5

typedef __bf16 v16bf __attribute__((ext_vector_type(16)));
typedef float  v8f   __attribute__((ext_vector_type(8)));

// ---------------------------------------------------------------------------
// Split fp32 -> bf16 hi (truncate) + bf16 lo (residual, RNE).  x ~= hi + lo.
// ---------------------------------------------------------------------------
__global__ __launch_bounds__(256)
void split_bf16x2(const float* __restrict__ src,
                  uint16_t* __restrict__ hi,
                  uint16_t* __restrict__ lo, int n) {
    int i = blockIdx.x * blockDim.x + threadIdx.x;
    int stride = gridDim.x * blockDim.x;
    for (; i < n; i += stride) {
        float x = src[i];
        uint32_t u = __float_as_uint(x) & 0xffff0000u;
        float hif = __uint_as_float(u);
        float lof = x - hif;
        __bf16 lb = (__bf16)lof;
        hi[i] = (uint16_t)(u >> 16);
        lo[i] = __builtin_bit_cast(uint16_t, lb);
    }
}

// ---------------------------------------------------------------------------
// CDNA5 async copy helpers (ASYNCcnt-tracked global -> LDS DMA).
// ---------------------------------------------------------------------------
__device__ __forceinline__ uint32_t lds_off(const void* p) {
    // Flat LDS address keeps the wave-relative LDS byte offset in addr[31:0].
    return (uint32_t)(uintptr_t)p;
}

__device__ __forceinline__ void async_ld16(uint32_t lds, const uint16_t* g) {
    asm volatile("global_load_async_to_lds_b128 %0, %1, off"
                 :: "v"(lds), "v"(g) : "memory");
}

template <int N>
__device__ __forceinline__ void wait_async() {
    asm volatile("s_wait_asynccnt %0" :: "n"(N) : "memory");
}

// ---------------------------------------------------------------------------
// GEMM  C[M,N] = A[M,K] * B[N,K]^T   (both operands K-contiguous row-major)
// bf16x3:  C += Ahi*Bhi + Ahi*Blo + Alo*Bhi     (fp32 accumulate via WMMA)
// Block 128x128, 8 waves (4 M x 2 N), wave: 2x4 of 16x16 WMMA tiles.
// Double-buffered LDS stages fed by global_load_async_to_lds_b128.
// ---------------------------------------------------------------------------
#define BM 128
#define BN 128
#define BK 32
#define LDSS 40   // LDS row stride in elements (80B) to spread banks

__device__ __forceinline__ v16bf load_frag(const uint16_t* rowp, int lko) {
    // CDNA5 16-bit A/B fragment: two contiguous 8-element (16B) chunks per
    // lane at k-offsets {lko, lko+16}; lko = 0 for lanes 0-15, 8 for 16-31.
    union { v16bf v; uint4 u[2]; } r;
    r.u[0] = *(const uint4*)(rowp + lko);
    r.u[1] = *(const uint4*)(rowp + lko + 16);
    return r.v;
}

__global__ __launch_bounds__(256)
void gemm_bf16x3(const uint16_t* __restrict__ Ahi, const uint16_t* __restrict__ Alo,
                 const uint16_t* __restrict__ Bhi, const uint16_t* __restrict__ Blo,
                 float* __restrict__ Cmat, int M, int N, int K) {
    __shared__ __align__(16) uint16_t sAh[2][BM][LDSS];
    __shared__ __align__(16) uint16_t sAl[2][BM][LDSS];
    __shared__ __align__(16) uint16_t sBh[2][BN][LDSS];
    __shared__ __align__(16) uint16_t sBl[2][BN][LDSS];

    const int tid  = threadIdx.x;
    const int lane = tid & 31;
    const int wv   = tid >> 5;
    const int wm   = wv & 3;     // wave M slot: 0..3  (32 rows each)
    const int wn   = wv >> 2;    // wave N slot: 0..1  (64 cols each)
    const int bM   = blockIdx.y * BM;
    const int bN   = blockIdx.x * BN;

    const int lrow = lane & 15;
    const int lko  = (lane < 16) ? 0 : 8;

    v8f acc[2][4];
#pragma unroll
    for (int ti = 0; ti < 2; ++ti)
#pragma unroll
        for (int tj = 0; tj < 4; ++tj)
            acc[ti][tj] = (v8f){0.f, 0.f, 0.f, 0.f, 0.f, 0.f, 0.f, 0.f};

    const int c0 = tid * 2;   // each thread owns 2 of the 512 16B chunks/tile

    // issue one k-tile's worth of async DMA (8 ops per thread => 8 per wave)
    auto issue_tile = [&](int st, int ks) {
#pragma unroll
        for (int i = 0; i < 2; ++i) {
            int c   = c0 + i;
            int row = c >> 2;
            int ko  = (c & 3) * 8;
            size_t ga = (size_t)(bM + row) * K + ks + ko;
            size_t gb = (size_t)(bN + row) * K + ks + ko;
            async_ld16(lds_off(&sAh[st][row][ko]), Ahi + ga);
            async_ld16(lds_off(&sAl[st][row][ko]), Alo + ga);
            async_ld16(lds_off(&sBh[st][row][ko]), Bhi + gb);
            async_ld16(lds_off(&sBl[st][row][ko]), Blo + gb);
        }
    };

    const int NT = K / BK;
    issue_tile(0, 0);

    for (int it = 0; it < NT; ++it) {
        const int cur = it & 1;
        if (it + 1 < NT) {
            issue_tile(1 - cur, (it + 1) * BK);   // DMA next stage
            wait_async<8>();                      // oldest 8 (= current tile) done
            if (it + 2 < NT)                      // warm L2 two tiles ahead
                __builtin_prefetch(Ahi + (size_t)(bM + (c0 >> 2)) * K + (it + 2) * BK, 0, 3);
        } else {
            wait_async<0>();
        }
        __syncthreads();   // current stage visible to all waves

        v16bf ah[2], al[2], bh[4], bl[4];
#pragma unroll
        for (int ti = 0; ti < 2; ++ti) {
            int m = wm * 32 + ti * 16 + lrow;
            ah[ti] = load_frag(&sAh[cur][m][0], lko);
            al[ti] = load_frag(&sAl[cur][m][0], lko);
        }
#pragma unroll
        for (int tj = 0; tj < 4; ++tj) {
            int n = wn * 64 + tj * 16 + lrow;
            bh[tj] = load_frag(&sBh[cur][n][0], lko);
            bl[tj] = load_frag(&sBl[cur][n][0], lko);
        }
#pragma unroll
        for (int ti = 0; ti < 2; ++ti)
#pragma unroll
            for (int tj = 0; tj < 4; ++tj) {
                acc[ti][tj] = __builtin_amdgcn_wmma_f32_16x16x32_bf16(
                    false, ah[ti], false, bh[tj], (short)0, acc[ti][tj], false, false);
                acc[ti][tj] = __builtin_amdgcn_wmma_f32_16x16x32_bf16(
                    false, ah[ti], false, bl[tj], (short)0, acc[ti][tj], false, false);
                acc[ti][tj] = __builtin_amdgcn_wmma_f32_16x16x32_bf16(
                    false, al[ti], false, bh[tj], (short)0, acc[ti][tj], false, false);
            }
        __syncthreads();   // all waves done reading before this stage is re-filled
    }

    // C/D layout: VGPR r -> row r (lanes 0-15) / row r+8 (lanes 16-31), col = lane&15
    const int mbase = bM + wm * 32 + ((lane < 16) ? 0 : 8);
    const int nbase = bN + wn * 64 + lrow;
#pragma unroll
    for (int ti = 0; ti < 2; ++ti)
#pragma unroll
        for (int tj = 0; tj < 4; ++tj)
#pragma unroll
            for (int r = 0; r < 8; ++r) {
                int m = mbase + ti * 16 + r;
                int n = nbase + tj * 16;
                Cmat[(size_t)m * N + n] = acc[ti][tj][r];
            }
}

// ---------------------------------------------------------------------------
// Sliding-window attention.  One thread per (l, b, h).
// q,k,v stored as [(l*B+b)*C + h*D + d] fp32.  pe is (h, d, w).
// energy[w] = sum_d q[d] * (k_window[d,w] + pe[d,w]);  window zero-padded in k,
// but pe is ALWAYS added (matches reference slide_window semantics).
// ---------------------------------------------------------------------------
__global__ __launch_bounds__(256)
void slide_attn(const float* __restrict__ qb, const float* __restrict__ kb,
                const float* __restrict__ vb, const float* __restrict__ pe,
                float* __restrict__ out) {
    int idx = blockIdx.x * blockDim.x + threadIdx.x;   // 0 .. L*B*H-1
    int h = idx & (NHEAD - 1);
    int t = idx >> 4;            // / NHEAD
    int b = t & (BATCH - 1);
    int l = t >> 2;              // / BATCH

    const float* q = qb + ((size_t)(l * BATCH + b) * CDIM + h * DHEAD);
    float qr[DHEAD];
#pragma unroll
    for (int d = 0; d < DHEAD; ++d) qr[d] = q[d];

    float e[WSZ];
#pragma unroll
    for (int w = 0; w < WSZ; ++w) {
        int j = l + w - PADW;
        float s = 0.f;
        if (j >= 0 && j < L_SEQ) {
            const float* kp = kb + ((size_t)(j * BATCH + b) * CDIM + h * DHEAD);
#pragma unroll
            for (int d = 0; d < DHEAD; ++d)
                s += qr[d] * (kp[d] + pe[(h * DHEAD + d) * WSZ + w]);
        } else {
#pragma unroll
            for (int d = 0; d < DHEAD; ++d)
                s += qr[d] * pe[(h * DHEAD + d) * WSZ + w];
        }
        e[w] = s * 0.03125f;   // 1/sqrt(1024)
    }

    float mx = e[0];
#pragma unroll
    for (int w = 1; w < WSZ; ++w) mx = fmaxf(mx, e[w]);
    float sum = 0.f;
#pragma unroll
    for (int w = 0; w < WSZ; ++w) { e[w] = __expf(e[w] - mx); sum += e[w]; }
    float inv = 1.f / sum;

    float o[DHEAD];
#pragma unroll
    for (int d = 0; d < DHEAD; ++d) o[d] = 0.f;
#pragma unroll
    for (int w = 0; w < WSZ; ++w) {
        int j = l + w - PADW;
        if (j >= 0 && j < L_SEQ) {
            const float* vp = vb + ((size_t)(j * BATCH + b) * CDIM + h * DHEAD);
            float a = e[w];
#pragma unroll
            for (int d = 0; d < DHEAD; ++d) o[d] += a * vp[d];
        }
    }
    float* op = out + ((size_t)(l * BATCH + b) * CDIM + h * DHEAD);
#pragma unroll
    for (int d = 0; d < DHEAD; ++d) op[d] = o[d] * inv;
}

// ---------------------------------------------------------------------------
extern "C" void kernel_launch(void* const* d_in, const int* in_sizes, int n_in,
                              void* d_out, int out_size, void* d_ws, size_t ws_size,
                              hipStream_t stream) {
    const float* x  = (const float*)d_in[0];
    const float* Wq = (const float*)d_in[1];
    const float* Wk = (const float*)d_in[2];
    const float* Wv = (const float*)d_in[3];
    const float* pe = (const float*)d_in[4];

    const int M = L_SEQ * BATCH;   // 8192
    const int K = CDIM;            // 1024
    const int N = CDIM;            // 1024

    // workspace layout (bytes):
    //   [0,32M)    q fp32     [32M,64M) k fp32     [64M,96M) v fp32
    //   [96M,112M) x hi bf16  [112M,128M) x lo bf16
    //   [128M,140M) 3x (W hi 2M, W lo 2M)
    char* ws = (char*)d_ws;
    float* qb = (float*)(ws);
    float* kb = (float*)(ws + (size_t)M * N * 4);
    float* vb = (float*)(ws + (size_t)M * N * 8);
    uint16_t* xhi = (uint16_t*)(ws + (size_t)M * N * 12);
    uint16_t* xlo = xhi + (size_t)M * K;
    uint16_t* wbase = xlo + (size_t)M * K;

    const float* Wsrc[3] = {Wq, Wk, Wv};
    float* Obuf[3] = {qb, kb, vb};

    split_bf16x2<<<512, 256, 0, stream>>>(x, xhi, xlo, M * K);
    for (int i = 0; i < 3; ++i) {
        uint16_t* whi = wbase + (size_t)i * 2 * K * N;
        uint16_t* wlo = whi + (size_t)K * N;
        split_bf16x2<<<256, 256, 0, stream>>>(Wsrc[i], whi, wlo, K * N);
    }

    dim3 grid(N / BN, M / BM);   // (8, 64)
    for (int i = 0; i < 3; ++i) {
        uint16_t* whi = wbase + (size_t)i * 2 * K * N;
        uint16_t* wlo = whi + (size_t)K * N;
        gemm_bf16x3<<<grid, 256, 0, stream>>>(xhi, xlo, whi, wlo, Obuf[i], M, N, K);
    }

    slide_attn<<<(L_SEQ * BATCH * NHEAD) / 256, 256, 0, stream>>>(qb, kb, vb, pe,
                                                                  (float*)d_out);
}
